// ContextualEncoder2_81793357185352
// MI455X (gfx1250) — matrix operand order repaired and checked by reading, not compiled, not run on hardware
//
#include <hip/hip_runtime.h>
#include <hip/hip_bf16.h>

// ---------------- problem constants ----------------
#define NB   16384      // buoys
#define TS   8          // time steps stored in obs
#define FD   64         // feature dim
#define HH   1024       // hidden
#define KX   1088       // input size = FD + HH
#define G3   3072       // 3*HH

typedef __attribute__((ext_vector_type(8)))  float  v8f;
typedef __attribute__((ext_vector_type(8)))  __bf16 bf16x8;
typedef __attribute__((ext_vector_type(16))) __bf16 bf16x16;

// ---------------- helpers ----------------
__device__ __forceinline__ __bf16 f2bf(float f) {
    union { float f; unsigned u; } in; in.f = f;
    unsigned r = in.u + 0x7FFFu + ((in.u >> 16) & 1u);   // round-to-nearest-even
    union { unsigned short s; __bf16 b; } out; out.s = (unsigned short)(r >> 16);
    return out.b;
}
__device__ __forceinline__ float sigm(float x) {
    return 1.0f / (1.0f + __expf(-x));
}
__device__ __forceinline__ float tanh_fast(float x) {
    float e = __expf(-2.0f * x);
    return (1.0f - e) / (1.0f + e);
}

// ---------------- small utility kernels ----------------
__global__ void f32_to_bf16_kernel(__bf16* __restrict__ dst,
                                   const float* __restrict__ src, int count) {
    int i = blockIdx.x * blockDim.x + threadIdx.x;
    if (i < count) dst[i] = f2bf(src[i]);
}

__global__ void zero_h_kernel(float* __restrict__ hF, __bf16* __restrict__ hb) {
    int i = blockIdx.x * blockDim.x + threadIdx.x;   // count = NB*HH
    if (i < NB * HH) { hF[i] = 0.0f; hb[i] = (__bf16)0.0f; }
}

// x = [obs[:,t,:], emb[ids]]  -> bf16, row-major NB x KX
__global__ void pack_obs_emb_kernel(__bf16* __restrict__ xb,
                                    const float* __restrict__ obs,
                                    const float* __restrict__ emb,
                                    const int* __restrict__ ids, int t) {
    size_t idx = (size_t)blockIdx.x * blockDim.x + threadIdx.x;
    if (idx >= (size_t)NB * KX) return;
    int n = (int)(idx / KX);
    int k = (int)(idx % KX);
    float v;
    if (k < FD) v = obs[((size_t)n * TS + t) * FD + k];
    else        v = emb[(size_t)ids[n] * HH + (k - FD)];
    xb[idx] = f2bf(v);
}

// x = [h_src_bf16 (HH), obs[:,t,:] (FD)] -> bf16, row-major NB x KX
__global__ void pack_h_obs_kernel(__bf16* __restrict__ xb,
                                  const __bf16* __restrict__ hsrc,
                                  const float* __restrict__ obs, int t) {
    size_t idx = (size_t)blockIdx.x * blockDim.x + threadIdx.x;
    if (idx >= (size_t)NB * KX) return;
    int n = (int)(idx / KX);
    int k = (int)(idx % KX);
    if (k < HH) xb[idx] = hsrc[(size_t)n * HH + k];
    else        xb[idx] = f2bf(obs[((size_t)n * TS + t) * FD + (k - HH)]);
}

// ---------------- fused GRU step (WMMA bf16) ----------------
// Block: 256 threads = 8 waves. Block tile: 128 rows x 32 cols of h_new.
// Wave w (waveM = w>>1, waveN = w&1) computes TWO 16x16 M-subtiles at
// m0 = by*128 + waveM*32 (+0 / +16) and one 16-wide column block at
// j0 = bx*32 + waveN*16. B (weight) fragments are reused across both
// M-subtiles -> 10 b128 loads per 6 WMMAs instead of 16.
__global__ void __launch_bounds__(256)
gru_step_kernel(const __bf16* __restrict__ xb,     // NB x KX
                const __bf16* __restrict__ hbin,   // NB x HH (bf16 shadow of h)
                const __bf16* __restrict__ WihB,   // G3 x KX
                const __bf16* __restrict__ WhhB,   // G3 x HH
                const float*  __restrict__ b_ih,   // G3
                const float*  __restrict__ b_hh,   // G3
                float*        __restrict__ hF,     // NB x HH (in: h_prev, out: h_new)
                __bf16*       __restrict__ hbout)  // NB x HH (bf16 of h_new)
{
    const int lane   = threadIdx.x & 31;
    const int wid    = threadIdx.x >> 5;
    const int laneLo = lane & 15;     // A: M offset; B: column; C: N
    const int laneHi = lane >> 4;     // A/B K-half select; C: M-half select

    const int m0 = blockIdx.y * 128 + (wid >> 1) * 32;   // rows m0..m0+31
    const int j0 = blockIdx.x * 32  + (wid & 1) * 16;    // cols j0..j0+15

    v8f accr0 = {}, accz0 = {}, accin0 = {}, acchn0 = {};
    v8f accr1 = {}, accz1 = {}, accin1 = {}, acchn1 = {};

    // ---- GEMM 1: x (NB x KX) * Wih^T, three gate column blocks ----
    {
        const __bf16* ap0 = xb + (size_t)(m0 + laneLo) * KX + laneHi * 16;
        const __bf16* ap1 = ap0 + (size_t)16 * KX;
        const __bf16* bp0 = WihB + ((size_t)(0 * HH + j0 + laneLo)) * KX + laneHi * 16;
        const __bf16* bp1 = WihB + ((size_t)(1 * HH + j0 + laneLo)) * KX + laneHi * 16;
        const __bf16* bp2 = WihB + ((size_t)(2 * HH + j0 + laneLo)) * KX + laneHi * 16;
#pragma unroll 2
        for (int kt = 0; kt < KX / 32; ++kt) {
            const int kk = kt * 32;
            union { bf16x16 v; bf16x8 h[2]; } a0, a1;
            a0.v = *(const bf16x16*)(ap0 + kk);
            a1.v = *(const bf16x16*)(ap1 + kk);
            bf16x16 b0 = *(const bf16x16*)(bp0 + kk);
            bf16x16 b1 = *(const bf16x16*)(bp1 + kk);
            bf16x16 b2 = *(const bf16x16*)(bp2 + kk);
            accr0  = __builtin_amdgcn_wmma_f32_16x16x32_bf16(false, a0.v, false, b0,
                                                             (short)0, accr0,  false, false);
            accr1  = __builtin_amdgcn_wmma_f32_16x16x32_bf16(false, a1.v, false, b0,
                                                             (short)0, accr1,  false, false);
            accz0  = __builtin_amdgcn_wmma_f32_16x16x32_bf16(false, a0.v, false, b1,
                                                             (short)0, accz0,  false, false);
            accz1  = __builtin_amdgcn_wmma_f32_16x16x32_bf16(false, a1.v, false, b1,
                                                             (short)0, accz1,  false, false);
            accin0 = __builtin_amdgcn_wmma_f32_16x16x32_bf16(false, a0.v, false, b2,
                                                             (short)0, accin0, false, false);
            accin1 = __builtin_amdgcn_wmma_f32_16x16x32_bf16(false, a1.v, false, b2,
                                                             (short)0, accin1, false, false);
        }
    }

    // ---- GEMM 2: h (NB x HH) * Whh^T; r,z accumulate on top, n separate ----
    {
        const __bf16* ap0 = hbin + (size_t)(m0 + laneLo) * HH + laneHi * 16;
        const __bf16* ap1 = ap0 + (size_t)16 * HH;
        const __bf16* bp0 = WhhB + ((size_t)(0 * HH + j0 + laneLo)) * HH + laneHi * 16;
        const __bf16* bp1 = WhhB + ((size_t)(1 * HH + j0 + laneLo)) * HH + laneHi * 16;
        const __bf16* bp2 = WhhB + ((size_t)(2 * HH + j0 + laneLo)) * HH + laneHi * 16;
#pragma unroll 2
        for (int kt = 0; kt < HH / 32; ++kt) {
            const int kk = kt * 32;
            union { bf16x16 v; bf16x8 h[2]; } a0, a1;
            a0.v = *(const bf16x16*)(ap0 + kk);
            a1.v = *(const bf16x16*)(ap1 + kk);
            bf16x16 b0 = *(const bf16x16*)(bp0 + kk);
            bf16x16 b1 = *(const bf16x16*)(bp1 + kk);
            bf16x16 b2 = *(const bf16x16*)(bp2 + kk);
            accr0  = __builtin_amdgcn_wmma_f32_16x16x32_bf16(false, a0.v, false, b0,
                                                             (short)0, accr0,  false, false);
            accr1  = __builtin_amdgcn_wmma_f32_16x16x32_bf16(false, a1.v, false, b0,
                                                             (short)0, accr1,  false, false);
            accz0  = __builtin_amdgcn_wmma_f32_16x16x32_bf16(false, a0.v, false, b1,
                                                             (short)0, accz0,  false, false);
            accz1  = __builtin_amdgcn_wmma_f32_16x16x32_bf16(false, a1.v, false, b1,
                                                             (short)0, accz1,  false, false);
            acchn0 = __builtin_amdgcn_wmma_f32_16x16x32_bf16(false, a0.v, false, b2,
                                                             (short)0, acchn0, false, false);
            acchn1 = __builtin_amdgcn_wmma_f32_16x16x32_bf16(false, a1.v, false, b2,
                                                             (short)0, acchn1, false, false);
        }
    }

    // ---- gate fusion epilogue ----
    // C/D layout: VGPR i holds M = mbase + i + laneHi*8, N = j0 + laneLo.
    const int jcol = j0 + laneLo;
    const float bir = b_ih[jcol],          bhr = b_hh[jcol];
    const float biz = b_ih[jcol + HH],     bhz = b_hh[jcol + HH];
    const float bin = b_ih[jcol + 2 * HH], bhn = b_hh[jcol + 2 * HH];

#pragma unroll
    for (int s = 0; s < 2; ++s) {
        const int mbase = m0 + s * 16;
        const v8f ar = s ? accr1 : accr0;
        const v8f az = s ? accz1 : accz0;
        const v8f ai = s ? accin1 : accin0;
        const v8f ah = s ? acchn1 : acchn0;
#pragma unroll
        for (int i = 0; i < 8; ++i) {
            const int m = mbase + i + laneHi * 8;
            const size_t hi = (size_t)m * HH + jcol;
            float r  = sigm(ar[i] + bir + bhr);
            float z  = sigm(az[i] + biz + bhz);
            float nn = tanh_fast(ai[i] + bin + r * (ah[i] + bhn));
            float hp = hF[hi];
            float hn = (1.0f - z) * nn + z * hp;
            hF[hi]    = hn;
            hbout[hi] = f2bf(hn);
        }
    }
}

// ---------------- host side ----------------
extern "C" void kernel_launch(void* const* d_in, const int* in_sizes, int n_in,
                              void* d_out, int out_size, void* d_ws, size_t ws_size,
                              hipStream_t stream) {
    (void)in_sizes; (void)n_in; (void)out_size; (void)ws_size;

    const float* obs  = (const float*)d_in[0];   // (NB, TS, FD)
    const int*   ids  = (const int*)  d_in[1];   // (NB,)
    const float* emb  = (const float*)d_in[2];   // (100, HH)
    const float* W_ih = (const float*)d_in[3];   // (G3, KX)
    const float* W_hh = (const float*)d_in[4];   // (G3, HH)
    const float* b_ih = (const float*)d_in[5];   // (G3,)
    const float* b_hh = (const float*)d_in[6];   // (G3,)
    float* hF = (float*)d_out;                   // running h lives in d_out (NB x HH)

    // workspace layout (bytes, all 32B aligned)
    char* ws = (char*)d_ws;
    const size_t SZ_WIHB = (size_t)G3 * KX * 2;      //  6,684,672
    const size_t SZ_WHHB = (size_t)G3 * HH * 2;      //  6,291,456
    const size_t SZ_XB   = (size_t)NB * KX * 2;      // 35,651,584
    const size_t SZ_HB   = (size_t)NB * HH * 2;      // 33,554,432
    __bf16* WihB  = (__bf16*)(ws);
    __bf16* WhhB  = (__bf16*)(ws + SZ_WIHB);
    __bf16* xb    = (__bf16*)(ws + SZ_WIHB + SZ_WHHB);
    __bf16* hb0   = (__bf16*)(ws + SZ_WIHB + SZ_WHHB + SZ_XB);
    __bf16* hb1   = (__bf16*)(ws + SZ_WIHB + SZ_WHHB + SZ_XB + SZ_HB);
    __bf16* out0b = (__bf16*)(ws + SZ_WIHB + SZ_WHHB + SZ_XB + 2 * SZ_HB);

    const int TPB = 256;
    // 1) convert weights to bf16 (every call: no static state allowed)
    {
        int c1 = G3 * KX, c2 = G3 * HH;
        f32_to_bf16_kernel<<<(c1 + TPB - 1) / TPB, TPB, 0, stream>>>(WihB, W_ih, c1);
        f32_to_bf16_kernel<<<(c2 + TPB - 1) / TPB, TPB, 0, stream>>>(WhhB, W_hh, c2);
    }
    // 2) h = 0 (f32 in d_out, bf16 shadow)
    zero_h_kernel<<<(NB * HH + TPB - 1) / TPB, TPB, 0, stream>>>(hF, hb0);

    const size_t nPack = (size_t)NB * KX;
    const int packBlocks = (int)((nPack + TPB - 1) / TPB);
    dim3 gruGrid(HH / 32, NB / 128);   // 32 x 128 blocks, 256 thr each

    // ---- step 1: x=[obs0, emb]; h: hb0 -> hb1 ----
    pack_obs_emb_kernel<<<packBlocks, TPB, 0, stream>>>(xb, obs, emb, ids, 0);
    gru_step_kernel<<<gruGrid, TPB, 0, stream>>>(xb, hb0, WihB, WhhB, b_ih, b_hh, hF, hb1);
    // save outs[0] (needed for step 4 input)
    hipMemcpyAsync(out0b, hb1, SZ_HB, hipMemcpyDeviceToDevice, stream);

    // ---- step 2: x=[obs1, emb]; hb1 -> hb0 ----
    pack_obs_emb_kernel<<<packBlocks, TPB, 0, stream>>>(xb, obs, emb, ids, 1);
    gru_step_kernel<<<gruGrid, TPB, 0, stream>>>(xb, hb1, WihB, WhhB, b_ih, b_hh, hF, hb0);

    // ---- step 3: x=[obs2, emb]; hb0 -> hb1 ----
    pack_obs_emb_kernel<<<packBlocks, TPB, 0, stream>>>(xb, obs, emb, ids, 2);
    gru_step_kernel<<<gruGrid, TPB, 0, stream>>>(xb, hb0, WihB, WhhB, b_ih, b_hh, hF, hb1);

    // ---- step 4: x=[outs0, obs1]; hb1 -> hb0 ----
    pack_h_obs_kernel<<<packBlocks, TPB, 0, stream>>>(xb, out0b, obs, 1);
    gru_step_kernel<<<gruGrid, TPB, 0, stream>>>(xb, hb1, WihB, WhhB, b_ih, b_hh, hF, hb0);

    // ---- step 5: x=[h_step4, obs2]; hb0 -> hb1 ----
    pack_h_obs_kernel<<<packBlocks, TPB, 0, stream>>>(xb, hb0, obs, 2);
    gru_step_kernel<<<gruGrid, TPB, 0, stream>>>(xb, hb0, WihB, WhhB, b_ih, b_hh, hF, hb1);
    // final h (f32) already lives in d_out
}
